// Relation_11836929868697
// MI455X (gfx1250) — matrix operand order repaired and checked by reading, not compiled, not run on hardware
//
#include <hip/hip_runtime.h>
#include <hip/hip_bf16.h>

// ---------------------------------------------------------------------------
// Relation module for MI455X (gfx1250, wave32, WMMA).
// B=2, N=384, HIDDEN=512, D_REL=256, NUM_HEAD=8, HEAD_DIM=64.
// The per-(b,i,j) MLP chain runs on v_wmma_f32_16x16x32_f16 (f32 accum);
// rel [B,N,N,256] (302MB) is fused in LDS and never touches HBM.
// Each wave owns a 32-row tile (two 16-row M-tiles) so every B fragment
// feeds two WMMAs; LN stats come from accumulator registers (shfl butterfly)
// so all LDS staging is f16 -> ~31KB/block -> ~10 waves/WGP for HBM latency
// cover while streaming edge.
// ---------------------------------------------------------------------------

#define BATCH 2
#define NN    384
#define HID   512
#define RDIM  256
#define NH    8
#define HD    64
#define JT32  (NN / 32)          // 12 j-tiles of 32 per row
#define EPSV  1e-5f

typedef __attribute__((ext_vector_type(16))) _Float16 v16h;
typedef __attribute__((ext_vector_type(8)))  _Float16 v8h;
typedef __attribute__((ext_vector_type(8)))  float    v8f;

// ---------------------------------------------------------------------------
// Fragment loaders for v_wmma_f32_16x16x32_f16 (CDNA5 ISA):
//   A 16x32: lanes 0-15 -> M=lane, K={0..7,16..23}; lanes 16-31 -> K={8..15,24..31}
//   B 32x16: lanes 0-15 -> N=lane, K={0..7,16..23}; lanes 16-31 -> K={8..15,24..31}
//   C/D f32: lane L -> N=L&15, VGPR r -> M = r + 8*(L>=16)
// ---------------------------------------------------------------------------
__device__ __forceinline__ v16h load_a_frag(const _Float16* A, int lda, int kb) {
  const int lane = threadIdx.x & 31;
  const int rc = lane & 15, hi = lane >> 4;
  union U { v16h v; v8h h[2]; } a;
  const _Float16* ap = A + rc * lda + kb * 32 + hi * 8;
  a.h[0] = *(const v8h*)(ap);        // K = kb*32 + hi*8  + 0..7
  a.h[1] = *(const v8h*)(ap + 16);   // K = kb*32 + 16 + hi*8 + 0..7
  return a.v;
}
__device__ __forceinline__ v16h load_b_frag(const _Float16* B, int ldb, int kb) {
  const int lane = threadIdx.x & 31;
  const int rc = lane & 15, hi = lane >> 4;
  union U { v16h v; v8h h[2]; } b;
  const _Float16* bp = B + rc * ldb + kb * 32 + hi * 8;
  b.h[0] = *(const v8h*)(bp);        // K = kb*32 + hi*8  + 0..7
  b.h[1] = *(const v8h*)(bp + 16);   // K = kb*32 + 16 + hi*8 + 0..7
  return b.v;
}
#define WMMA(Af, Bf, Cacc) \
  __builtin_amdgcn_wmma_f32_16x16x32_f16(false, (Af), false, (Bf), (short)0, (Cacc), false, false)

__device__ __forceinline__ float warp_sum(float v) {
  #pragma unroll
  for (int off = 16; off > 0; off >>= 1) v += __shfl_xor(v, off, 32);
  return v;
}
__device__ __forceinline__ float warp_max(float v) {
  #pragma unroll
  for (int off = 16; off > 0; off >>= 1) v = fmaxf(v, __shfl_xor(v, off, 32));
  return v;
}
// reduce across the 16 lanes that share the same hi-half (masks stay in-group)
__device__ __forceinline__ float group16_sum(float v) {
  v += __shfl_xor(v, 8, 32);
  v += __shfl_xor(v, 4, 32);
  v += __shfl_xor(v, 2, 32);
  v += __shfl_xor(v, 1, 32);
  return v;
}

// ---------------------------------------------------------------------------
// Kernel P: transpose + f16-convert weights for WMMA B operands.
//   W1 [256,128] -> W1t[n=128][k=256]   W2 [128,64] -> W2t[64][128]
//   W3 [64,8]    -> W3t[16][64] (rows 8..15 zero)   Wo [512,512] -> Wot[512][512]
// ---------------------------------------------------------------------------
__global__ __launch_bounds__(256) void k_prep(
    const float* __restrict__ W1, const float* __restrict__ W2,
    const float* __restrict__ W3, const float* __restrict__ Wo,
    _Float16* __restrict__ W1t, _Float16* __restrict__ W2t,
    _Float16* __restrict__ W3t, _Float16* __restrict__ Wot) {
  const int idx = blockIdx.x * 256 + threadIdx.x;
  const int n1 = 128 * 256, n2 = 64 * 128, n3 = 16 * 64, n4 = 512 * 512;
  if (idx < n1) {
    int n = idx / 256, kk = idx % 256;
    W1t[idx] = (_Float16)W1[kk * 128 + n];
  } else if (idx < n1 + n2) {
    int o = idx - n1; int n = o / 128, kk = o % 128;
    W2t[o] = (_Float16)W2[kk * 64 + n];
  } else if (idx < n1 + n2 + n3) {
    int o = idx - n1 - n2; int n = o / 64, kk = o % 64;
    W3t[o] = (n < 8) ? (_Float16)W3[kk * 8 + n] : (_Float16)0.f;
  } else if (idx < n1 + n2 + n3 + n4) {
    int o = idx - n1 - n2 - n3; int n = o / 512, kk = o % 512;
    Wot[o] = (_Float16)Wo[kk * 512 + n];
  }
}

// ---------------------------------------------------------------------------
// Kernel 1: x = relu(LN0(node)); q = x@Wq+bq; k = x@Wk+bk; v = x@Wv+bv.
// 8 rows per block (one wave per row for LN); weights stay L2-resident.
// ---------------------------------------------------------------------------
__global__ __launch_bounds__(256) void k_ln_qkv(
    const float* __restrict__ node,
    const float* __restrict__ g0, const float* __restrict__ b0,
    const float* __restrict__ Wq, const float* __restrict__ bq,
    const float* __restrict__ Wk, const float* __restrict__ bk,
    const float* __restrict__ Wv, const float* __restrict__ bv,
    float* __restrict__ q, float* __restrict__ k, float* __restrict__ v) {
  __shared__ __align__(16) float sx[8][512];
  const int tid = threadIdx.x, warp = tid >> 5, lane = tid & 31;
  const int row = blockIdx.x * 8 + warp;

  const float4* n4 = (const float4*)(node + (size_t)row * HID);
  float4 xv[4];
  float s = 0.f, ss = 0.f;
  #pragma unroll
  for (int c = 0; c < 4; ++c) {
    xv[c] = n4[c * 32 + lane];
    s  += xv[c].x + xv[c].y + xv[c].z + xv[c].w;
    ss += xv[c].x * xv[c].x + xv[c].y * xv[c].y +
          xv[c].z * xv[c].z + xv[c].w * xv[c].w;
  }
  s = warp_sum(s); ss = warp_sum(ss);
  const float m   = s * (1.f / HID);
  const float inv = rsqrtf(ss * (1.f / HID) - m * m + EPSV);
  #pragma unroll
  for (int c = 0; c < 4; ++c) {
    const int v4i = c * 32 + lane;
    float4 gg = ((const float4*)g0)[v4i];
    float4 bb = ((const float4*)b0)[v4i];
    float4 o;
    o.x = fmaxf((xv[c].x - m) * inv * gg.x + bb.x, 0.f);
    o.y = fmaxf((xv[c].y - m) * inv * gg.y + bb.y, 0.f);
    o.z = fmaxf((xv[c].z - m) * inv * gg.z + bb.z, 0.f);
    o.w = fmaxf((xv[c].w - m) * inv * gg.w + bb.w, 0.f);
    *(float4*)&sx[warp][v4i * 4] = o;
  }
  __syncthreads();

  const int c = tid;  // output column 0..255
  float aq[8], ak[8], av0[8], av1[8];
  #pragma unroll
  for (int r = 0; r < 8; ++r) { aq[r] = ak[r] = av0[r] = av1[r] = 0.f; }
  for (int kk4 = 0; kk4 < HID / 4; ++kk4) {
    const int kk = kk4 * 4;
    float wq[4], wk[4], w0[4], w1[4];
    #pragma unroll
    for (int u = 0; u < 4; ++u) {
      wq[u] = Wq[(size_t)(kk + u) * RDIM + c];
      wk[u] = Wk[(size_t)(kk + u) * RDIM + c];
      w0[u] = Wv[(size_t)(kk + u) * HID + c];
      w1[u] = Wv[(size_t)(kk + u) * HID + 256 + c];
    }
    #pragma unroll
    for (int r = 0; r < 8; ++r) {
      const float4 x = *(const float4*)&sx[r][kk];   // ds_load_b128 broadcast
      aq[r]  += x.x * wq[0] + x.y * wq[1] + x.z * wq[2] + x.w * wq[3];
      ak[r]  += x.x * wk[0] + x.y * wk[1] + x.z * wk[2] + x.w * wk[3];
      av0[r] += x.x * w0[0] + x.y * w0[1] + x.z * w0[2] + x.w * w0[3];
      av1[r] += x.x * w1[0] + x.y * w1[1] + x.z * w1[2] + x.w * w1[3];
    }
  }
  const int rb = blockIdx.x * 8;
  #pragma unroll
  for (int r = 0; r < 8; ++r) {
    const size_t rw = rb + r;
    q[rw * RDIM + c]      = aq[r]  + bq[c];
    k[rw * RDIM + c]      = ak[r]  + bk[c];
    v[rw * HID + c]       = av0[r] + bv[c];
    v[rw * HID + 256 + c] = av1[r] + bv[256 + c];
  }
}

// ---------------------------------------------------------------------------
// Kernel 2: one wave per 32-point tile (fixed b,i; j0..j0+31, two M-tiles).
//   rel = q*k + edge + dir_emb -> LN1 -> WMMA @W1t -> relu
//   -> LN2 -> WMMA @W2t -> +last_relation -> last_rel (global)
//   -> relu -> LN3 -> WMMA @W3t -> att logits (+b3 +rel_mask) -> ws
// LN stats computed from f32 accumulators (register butterflies); LDS
// intermediates stored as raw f16 and normalized in place.
// ---------------------------------------------------------------------------
__global__ __launch_bounds__(32) void k_relation(
    const float* __restrict__ qb, const float* __restrict__ kb_,
    const float* __restrict__ edge, const float* __restrict__ last_relation,
    const int* __restrict__ drctn, const float* __restrict__ rel_mask,
    const float* __restrict__ dir_emb,
    const float* __restrict__ ln1g, const float* __restrict__ ln1b,
    const _Float16* __restrict__ W1t, const float* __restrict__ b1,
    const float* __restrict__ ln2g, const float* __restrict__ ln2b,
    const _Float16* __restrict__ W2t, const float* __restrict__ b2,
    const float* __restrict__ ln3g, const float* __restrict__ ln3b,
    const _Float16* __restrict__ W3t, const float* __restrict__ b3,
    float* __restrict__ out_lr, float* __restrict__ logits) {
  const int bid = blockIdx.x;
  const int jt  = bid % JT32;
  const int i   = (bid / JT32) % NN;
  const int b   = bid / (JT32 * NN);
  const int j0  = jt * 32;
  const int lane = threadIdx.x;
  const int col = lane & 15, hi = lane >> 4;

  __shared__ __align__(16) float    s_q[RDIM];                 // 1 KB
  __shared__ __align__(16) _Float16 s_relA[32][264];           // 16.5 KB
  __shared__ __align__(16) _Float16 s_h1A[32][136];            // 8.5 KB
  __shared__ __align__(16) _Float16 s_lrA[32][72];             // 4.5 KB
  __shared__ float s_mean[32], s_inv[32];

  {
    const float* qr = qb + ((size_t)(b * NN + i)) * RDIM;
    for (int f = lane; f < RDIM; f += 32) s_q[f] = qr[f];
  }

  // ---- phase 1: rel = q*k + edge + dir_emb (f32 in regs -> raw f16 LDS) ----
  const size_t rowbase = (size_t)(b * NN + i) * NN;
  for (int t = 0; t < 32; ++t) {
    const int j = j0 + t;
    const int dir = drctn[rowbase + j];
    const float4* k4 = (const float4*)(kb_ + (size_t)(b * NN + j) * RDIM);
    const float4* e4 = (const float4*)(edge + (rowbase + j) * RDIM);
    const float4* d4 = (const float4*)(dir_emb + (size_t)dir * RDIM);
    if (t < 31 && lane == 0)
      __builtin_prefetch(e4 + 64, 0, 1);  // next j's edge row
    float s = 0.f, ss = 0.f;
    #pragma unroll
    for (int c = 0; c < 2; ++c) {
      const int v4i = c * 32 + lane;
      float4 qq = ((const float4*)s_q)[v4i];
      float4 kk = k4[v4i], ee = e4[v4i], dd = d4[v4i];
      const float r0 = qq.x * kk.x + ee.x + dd.x;
      const float r1 = qq.y * kk.y + ee.y + dd.y;
      const float r2 = qq.z * kk.z + ee.z + dd.z;
      const float r3 = qq.w * kk.w + ee.w + dd.w;
      const int f = v4i * 4;
      s_relA[t][f + 0] = (_Float16)r0;
      s_relA[t][f + 1] = (_Float16)r1;
      s_relA[t][f + 2] = (_Float16)r2;
      s_relA[t][f + 3] = (_Float16)r3;
      s  += r0 + r1 + r2 + r3;
      ss += r0 * r0 + r1 * r1 + r2 * r2 + r3 * r3;
    }
    s = warp_sum(s); ss = warp_sum(ss);
    if (lane == 0) {
      const float mm = s * (1.f / RDIM);
      s_mean[t] = mm;
      s_inv[t]  = rsqrtf(ss * (1.f / RDIM) - mm * mm + EPSV);
    }
  }
  // LN1 in place on f16
  for (int idx = lane; idx < 32 * 64; idx += 32) {
    const int t = idx >> 6, v4i = idx & 63, f = v4i * 4;
    const float mm = s_mean[t], inv = s_inv[t];
    float4 gg = ((const float4*)ln1g)[v4i];
    float4 bb = ((const float4*)ln1b)[v4i];
    s_relA[t][f + 0] = (_Float16)(((float)s_relA[t][f + 0] - mm) * inv * gg.x + bb.x);
    s_relA[t][f + 1] = (_Float16)(((float)s_relA[t][f + 1] - mm) * inv * gg.y + bb.y);
    s_relA[t][f + 2] = (_Float16)(((float)s_relA[t][f + 2] - mm) * inv * gg.z + bb.z);
    s_relA[t][f + 3] = (_Float16)(((float)s_relA[t][f + 3] - mm) * inv * gg.w + bb.w);
  }

  // ---- GEMM1: [32x256] @ W1t -> relu -> raw f16 + register LN2 stats ----
  float ps0[8], ss0[8], ps1[8], ss1[8];
  #pragma unroll
  for (int r = 0; r < 8; ++r) { ps0[r] = ss0[r] = ps1[r] = ss1[r] = 0.f; }
  for (int nb = 0; nb < 8; ++nb) {
    v8f a0 = {0.f,0.f,0.f,0.f,0.f,0.f,0.f,0.f};
    v8f a1 = {0.f,0.f,0.f,0.f,0.f,0.f,0.f,0.f};
    #pragma unroll
    for (int kkb = 0; kkb < 8; ++kkb) {
      const v16h bf  = load_b_frag(W1t + (size_t)(nb * 16) * 256, 256, kkb);
      const v16h af0 = load_a_frag(&s_relA[0][0],  264, kkb);
      const v16h af1 = load_a_frag(&s_relA[16][0], 264, kkb);
      a0 = WMMA(af0, bf, a0);
      a1 = WMMA(af1, bf, a1);
    }
    const float bias = b1[nb * 16 + col];
    #pragma unroll
    for (int r = 0; r < 8; ++r) {
      const float h0 = fmaxf(a0[r] + bias, 0.f);
      const float h1 = fmaxf(a1[r] + bias, 0.f);
      s_h1A[r + hi * 8][nb * 16 + col]      = (_Float16)h0;
      s_h1A[16 + r + hi * 8][nb * 16 + col] = (_Float16)h1;
      ps0[r] += h0; ss0[r] += h0 * h0;
      ps1[r] += h1; ss1[r] += h1 * h1;
    }
  }
  #pragma unroll
  for (int r = 0; r < 8; ++r) {
    ps0[r] = group16_sum(ps0[r]); ss0[r] = group16_sum(ss0[r]);
    ps1[r] = group16_sum(ps1[r]); ss1[r] = group16_sum(ss1[r]);
  }
  if (col == 0) {
    #pragma unroll
    for (int r = 0; r < 8; ++r) {
      const int m0 = r + hi * 8, m1 = 16 + m0;
      float mm = ps0[r] * (1.f / 128.f);
      s_mean[m0] = mm;
      s_inv[m0]  = rsqrtf(ss0[r] * (1.f / 128.f) - mm * mm + EPSV);
      mm = ps1[r] * (1.f / 128.f);
      s_mean[m1] = mm;
      s_inv[m1]  = rsqrtf(ss1[r] * (1.f / 128.f) - mm * mm + EPSV);
    }
  }
  // LN2 in place
  for (int idx = lane; idx < 32 * 32; idx += 32) {
    const int t = idx >> 5, v4i = idx & 31, f = v4i * 4;
    const float mm = s_mean[t], inv = s_inv[t];
    float4 gg = ((const float4*)ln2g)[v4i];
    float4 bb = ((const float4*)ln2b)[v4i];
    s_h1A[t][f + 0] = (_Float16)(((float)s_h1A[t][f + 0] - mm) * inv * gg.x + bb.x);
    s_h1A[t][f + 1] = (_Float16)(((float)s_h1A[t][f + 1] - mm) * inv * gg.y + bb.y);
    s_h1A[t][f + 2] = (_Float16)(((float)s_h1A[t][f + 2] - mm) * inv * gg.z + bb.z);
    s_h1A[t][f + 3] = (_Float16)(((float)s_h1A[t][f + 3] - mm) * inv * gg.w + bb.w);
  }

  // ---- GEMM2 + b2 + last_relation -> last_rel (global) + LN3 stats ----
  #pragma unroll
  for (int r = 0; r < 8; ++r) { ps0[r] = ss0[r] = ps1[r] = ss1[r] = 0.f; }
  const size_t lrbase = (rowbase + j0) * HD;
  for (int nb = 0; nb < 4; ++nb) {
    v8f a0 = {0.f,0.f,0.f,0.f,0.f,0.f,0.f,0.f};
    v8f a1 = {0.f,0.f,0.f,0.f,0.f,0.f,0.f,0.f};
    #pragma unroll
    for (int kkb = 0; kkb < 4; ++kkb) {
      const v16h bf  = load_b_frag(W2t + (size_t)(nb * 16) * 128, 128, kkb);
      const v16h af0 = load_a_frag(&s_h1A[0][0],  136, kkb);
      const v16h af1 = load_a_frag(&s_h1A[16][0], 136, kkb);
      a0 = WMMA(af0, bf, a0);
      a1 = WMMA(af1, bf, a1);
    }
    const int f = nb * 16 + col;
    const float bias = b2[f];
    #pragma unroll
    for (int r = 0; r < 8; ++r) {
      const int m0 = r + hi * 8, m1 = 16 + m0;
      const float lr0 = a0[r] + bias + last_relation[lrbase + (size_t)m0 * HD + f];
      const float lr1 = a1[r] + bias + last_relation[lrbase + (size_t)m1 * HD + f];
      out_lr[lrbase + (size_t)m0 * HD + f] = lr0;
      out_lr[lrbase + (size_t)m1 * HD + f] = lr1;
      const float a0r = fmaxf(lr0, 0.f), a1r = fmaxf(lr1, 0.f);
      s_lrA[m0][f] = (_Float16)a0r;
      s_lrA[m1][f] = (_Float16)a1r;
      ps0[r] += a0r; ss0[r] += a0r * a0r;
      ps1[r] += a1r; ss1[r] += a1r * a1r;
    }
  }
  #pragma unroll
  for (int r = 0; r < 8; ++r) {
    ps0[r] = group16_sum(ps0[r]); ss0[r] = group16_sum(ss0[r]);
    ps1[r] = group16_sum(ps1[r]); ss1[r] = group16_sum(ss1[r]);
  }
  if (col == 0) {
    #pragma unroll
    for (int r = 0; r < 8; ++r) {
      const int m0 = r + hi * 8, m1 = 16 + m0;
      float mm = ps0[r] * (1.f / 64.f);
      s_mean[m0] = mm;
      s_inv[m0]  = rsqrtf(ss0[r] * (1.f / 64.f) - mm * mm + EPSV);
      mm = ps1[r] * (1.f / 64.f);
      s_mean[m1] = mm;
      s_inv[m1]  = rsqrtf(ss1[r] * (1.f / 64.f) - mm * mm + EPSV);
    }
  }
  // LN3 in place (values already relu'd)
  for (int idx = lane; idx < 32 * 64; idx += 32) {
    const int t = idx >> 6, f = idx & 63;
    s_lrA[t][f] = (_Float16)(((float)s_lrA[t][f] - s_mean[t]) * s_inv[t] * ln3g[f] + ln3b[f]);
  }

  // ---- GEMM3: [32x64] @ W3t[16x64] (cols 8..15 zero) -> att logits ----
  {
    v8f a0 = {0.f,0.f,0.f,0.f,0.f,0.f,0.f,0.f};
    v8f a1 = {0.f,0.f,0.f,0.f,0.f,0.f,0.f,0.f};
    #pragma unroll
    for (int kkb = 0; kkb < 2; ++kkb) {
      const v16h bf  = load_b_frag(W3t, 64, kkb);
      const v16h af0 = load_a_frag(&s_lrA[0][0],  72, kkb);
      const v16h af1 = load_a_frag(&s_lrA[16][0], 72, kkb);
      a0 = WMMA(af0, bf, a0);
      a1 = WMMA(af1, bf, a1);
    }
    if (col < NH) {
      const float bias = b3[col];
      #pragma unroll
      for (int r = 0; r < 8; ++r) {
        const int m0 = r + hi * 8, m1 = 16 + m0;
        const size_t o0 = (rowbase + j0 + m0) * NH + col;
        const size_t o1 = (rowbase + j0 + m1) * NH + col;
        logits[o0] = a0[r] + bias + rel_mask[o0];
        logits[o1] = a1[r] + bias + rel_mask[o1];
      }
    }
  }
}

// ---------------------------------------------------------------------------
// Kernel 3: softmax over j (per head) + att@v. One block per (b,i).
// ---------------------------------------------------------------------------
__global__ __launch_bounds__(256) void k_softmax_av(
    const float* __restrict__ logits, const float* __restrict__ v,
    float* __restrict__ attout) {
  __shared__ float sa[NN * NH];  // 12 KB
  const int bi = blockIdx.x;     // b*N + i
  const int b  = bi / NN;
  const int tid = threadIdx.x, head = tid >> 5, lane = tid & 31;

  const float* lrow = logits + (size_t)bi * NN * NH;
  for (int idx = tid; idx < NN * NH; idx += 256) sa[idx] = lrow[idx];
  __syncthreads();

  float mx = -1e30f;
  for (int j = lane; j < NN; j += 32) mx = fmaxf(mx, sa[j * NH + head]);
  mx = warp_max(mx);
  float s = 0.f;
  for (int j = lane; j < NN; j += 32) {
    const float e = __expf(sa[j * NH + head] - mx);
    sa[j * NH + head] = e;
    s += e;
  }
  s = warp_sum(s);
  const float inv = 1.f / s;
  for (int j = lane; j < NN; j += 32) sa[j * NH + head] *= inv;
  __syncthreads();

  const float* vb = v + (size_t)b * NN * HID;
  for (int o = tid; o < HID; o += 256) {
    const int h = o >> 6;  // o = h*64 + d
    float acc = 0.f;
    for (int j = 0; j < NN; ++j) acc += sa[j * NH + h] * vb[(size_t)j * HID + o];
    attout[(size_t)bi * HID + o] = acc;
  }
}

// ---------------------------------------------------------------------------
// Kernel 4: out = attout @ Wo + bo via WMMA (M=768 in 16-row tiles, K=N=512).
// ---------------------------------------------------------------------------
__global__ __launch_bounds__(32) void k_out_gemm(
    const float* __restrict__ attout, const _Float16* __restrict__ Wot,
    const float* __restrict__ bo, float* __restrict__ out) {
  __shared__ __align__(16) _Float16 sA[16][520];
  const int lane = threadIdx.x;
  const int row0 = blockIdx.x * 16;

  for (int idx = lane; idx < 16 * 128; idx += 32) {
    const int t = idx >> 7, v4i = idx & 127, f = v4i * 4;
    float4 x = ((const float4*)(attout + (size_t)(row0 + t) * HID))[v4i];
    sA[t][f + 0] = (_Float16)x.x;
    sA[t][f + 1] = (_Float16)x.y;
    sA[t][f + 2] = (_Float16)x.z;
    sA[t][f + 3] = (_Float16)x.w;
  }

  const int col = lane & 15, hi = lane >> 4;
  for (int nb = 0; nb < 32; ++nb) {
    v8f acc = {0.f,0.f,0.f,0.f,0.f,0.f,0.f,0.f};
    #pragma unroll
    for (int kkb = 0; kkb < 16; ++kkb) {
      const v16h bf = load_b_frag(Wot + (size_t)(nb * 16) * 512, 512, kkb);
      const v16h af = load_a_frag(&sA[0][0], 520, kkb);
      acc = WMMA(af, bf, acc);
    }
    const float bias = bo[nb * 16 + col];
    #pragma unroll
    for (int r = 0; r < 8; ++r)
      out[(size_t)(row0 + r + hi * 8) * HID + nb * 16 + col] = acc[r] + bias;
  }
}

// ---------------------------------------------------------------------------
extern "C" void kernel_launch(void* const* d_in, const int* in_sizes, int n_in,
                              void* d_out, int out_size, void* d_ws, size_t ws_size,
                              hipStream_t stream) {
  (void)in_sizes; (void)n_in; (void)out_size; (void)ws_size;
  const float* node   = (const float*)d_in[0];
  const float* edge   = (const float*)d_in[1];
  const float* lastr  = (const float*)d_in[2];
  const int*   drctn  = (const int*)d_in[3];
  const float* relmsk = (const float*)d_in[4];
  const float* ln0g = (const float*)d_in[5],  *ln0b = (const float*)d_in[6];
  const float* Wq = (const float*)d_in[7],    *bq = (const float*)d_in[8];
  const float* Wk = (const float*)d_in[9],    *bk = (const float*)d_in[10];
  const float* Wv = (const float*)d_in[11],   *bv = (const float*)d_in[12];
  const float* ln1g = (const float*)d_in[13], *ln1b = (const float*)d_in[14];
  const float* W1 = (const float*)d_in[15],   *b1 = (const float*)d_in[16];
  const float* ln2g = (const float*)d_in[17], *ln2b = (const float*)d_in[18];
  const float* W2 = (const float*)d_in[19],   *b2 = (const float*)d_in[20];
  const float* dire = (const float*)d_in[21];
  const float* ln3g = (const float*)d_in[22], *ln3b = (const float*)d_in[23];
  const float* W3 = (const float*)d_in[24],   *b3 = (const float*)d_in[25];
  const float* Wo = (const float*)d_in[26],   *bo = (const float*)d_in[27];

  float* out    = (float*)d_out;                       // [B,N,512]
  float* out_lr = out + (size_t)BATCH * NN * HID;      // [B,N,N,64]

  // workspace carve-up (~14.2 MB)
  char* w = (char*)d_ws;
  float* ws_q   = (float*)w; w += (size_t)BATCH * NN * RDIM * 4;
  float* ws_k   = (float*)w; w += (size_t)BATCH * NN * RDIM * 4;
  float* ws_v   = (float*)w; w += (size_t)BATCH * NN * HID * 4;
  float* ws_lg  = (float*)w; w += (size_t)BATCH * NN * NN * NH * 4;
  float* ws_ao  = (float*)w; w += (size_t)BATCH * NN * HID * 4;
  _Float16* W1t = (_Float16*)w; w += 128 * 256 * 2;
  _Float16* W2t = (_Float16*)w; w += 64 * 128 * 2;
  _Float16* W3t = (_Float16*)w; w += 16 * 64 * 2;
  _Float16* Wot = (_Float16*)w; w += 512 * 512 * 2;

  {
    const int total = 128 * 256 + 64 * 128 + 16 * 64 + 512 * 512;
    k_prep<<<(total + 255) / 256, 256, 0, stream>>>(W1, W2, W3, Wo,
                                                    W1t, W2t, W3t, Wot);
  }
  k_ln_qkv<<<(BATCH * NN) / 8, 256, 0, stream>>>(node, ln0g, ln0b,
                                                 Wq, bq, Wk, bk, Wv, bv,
                                                 ws_q, ws_k, ws_v);
  k_relation<<<BATCH * NN * JT32, 32, 0, stream>>>(
      ws_q, ws_k, edge, lastr, drctn, relmsk, dire,
      ln1g, ln1b, W1t, b1, ln2g, ln2b, W2t, b2, ln3g, ln3b, W3t, b3,
      out_lr, ws_lg);
  k_softmax_av<<<BATCH * NN, 256, 0, stream>>>(ws_lg, ws_v, ws_ao);
  k_out_gemm<<<(BATCH * NN) / 16, 32, 0, stream>>>(ws_ao, Wot, bo, out);
}
